// ClassicalSelfAttention_65481071407606
// MI455X (gfx1250) — matrix operand order repaired and checked by reading, not compiled, not run on hardware
//
#include <hip/hip_runtime.h>
#include <hip/hip_bf16.h>
#include <math.h>

// ---------------------------------------------------------------------------
// ClassicalSelfAttention for MI455X (gfx1250, wave32, WMMA)
//   Q = X Wq / sqrt(D); K = X Wk; out = softmax(Q K^T) X
// bf16x3 emulated-fp32 GEMMs on v_wmma_f32_16x16x32_bf16, flash-style
// online softmax so the 8192x8192 score matrix never touches HBM.
// ---------------------------------------------------------------------------

constexpr int EMBED = 512;
constexpr int SEQ   = 8192;
constexpr int KC    = 128;   // keys per flash iteration = 8 waves * 16

typedef __attribute__((ext_vector_type(16))) __bf16 v16bf;
typedef __attribute__((ext_vector_type(8)))  float  v8f;

#define DEV __device__ __forceinline__

// A-matrix 16x32 bf16 fragment (ISA 7.12.2):
//  lanes 0-15 (M=lane): elems 0..7 -> K=kb+0..7,  elems 8..15 -> K=kb+16..23
//  lanes16-31 (M=lane-16): elems -> K=kb+8..15 and kb+24..31
DEV v16bf load_a_frag(const __bf16* row, int kb, int hih) {
  union { v16bf v; float4 f[2]; } u;
  u.f[0] = *(const float4*)(row + kb + hih * 8);
  u.f[1] = *(const float4*)(row + kb + 16 + hih * 8);
  return u.v;
}

// B-matrix 32x16 bf16 fragment (same K order as A): lane holds column N=lane&15,
// lanes 0-15 cover K=kb+0..7 and kb+16..23, lanes 16-31 cover K=kb+8..15 and kb+24..31.
DEV v16bf load_b_frag(const __bf16* row, int kb, int hih) {
  union { v16bf v; float4 f[2]; } u;
  const __bf16* p = row + kb + hih * 8;
  u.f[0] = *(const float4*)(p);
  u.f[1] = *(const float4*)(p + 16);
  return u.v;
}

DEV v8f bfmma(v16bf a, v16bf b, v8f c) {
  return __builtin_amdgcn_wmma_f32_16x16x32_bf16(
      /*neg_a=*/false, a, /*neg_b=*/false, b,
      /*c_mod=*/(short)0, c, /*reuse_a=*/false, /*reuse_b=*/false);
}

// ---------------------------------------------------------------------------
// Kernel 1: fp32 -> split bf16 (hi + lo) with a transposed copy.
// Fixed 512 columns (both X[N,512] and W[512,512]).
// ---------------------------------------------------------------------------
__global__ __launch_bounds__(256) void csa_split512(
    const float* __restrict__ src,
    __bf16* __restrict__ hi,  __bf16* __restrict__ lo,
    __bf16* __restrict__ thi, __bf16* __restrict__ tlo, int rows) {
  size_t idx = (size_t)blockIdx.x * 256 + threadIdx.x;
  if (idx >= (size_t)rows * EMBED) return;
  int r = (int)(idx >> 9);
  int c = (int)(idx & 511);
  float v = src[idx];
  __bf16 h = (__bf16)v;
  __bf16 l = (__bf16)(v - (float)h);
  hi[idx] = h; lo[idx] = l;
  size_t t = (size_t)c * rows + r;
  thi[t] = h; tlo[t] = l;
}

// ---------------------------------------------------------------------------
// Kernel 2: Q = scale * X Wq, K = X Wk (bf16x3 WMMA), re-split to bf16 hi/lo.
// grid = (4, SEQ/16, 2); 8 waves per block, one 16x16 output tile per wave.
// ---------------------------------------------------------------------------
__global__ __launch_bounds__(256) void csa_qk_proj(
    const __bf16* __restrict__ Xhi,   const __bf16* __restrict__ Xlo,
    const __bf16* __restrict__ Wqthi, const __bf16* __restrict__ Wqtlo,
    const __bf16* __restrict__ Wkthi, const __bf16* __restrict__ Wktlo,
    __bf16* __restrict__ Qhi, __bf16* __restrict__ Qlo,
    __bf16* __restrict__ Khi, __bf16* __restrict__ Klo) {
  const int wave = threadIdx.x >> 5;
  const int lane = threadIdx.x & 31;
  const int mrow = lane & 15;
  const int hih  = lane >> 4;
  const int ntile = blockIdx.x * 8 + wave;   // 0..31
  const int mtile = blockIdx.y;              // 0..511
  const int isK   = blockIdx.z;

  const __bf16* Wthi = isK ? Wkthi : Wqthi;
  const __bf16* Wtlo = isK ? Wktlo : Wqtlo;
  __bf16* Ohi = isK ? Khi : Qhi;
  __bf16* Olo = isK ? Klo : Qlo;

  const __bf16* aH = Xhi + (size_t)(mtile * 16 + mrow) * EMBED;
  const __bf16* aL = Xlo + (size_t)(mtile * 16 + mrow) * EMBED;
  const __bf16* bH = Wthi + (size_t)(ntile * 16 + mrow) * EMBED;  // W^T row = out col
  const __bf16* bL = Wtlo + (size_t)(ntile * 16 + mrow) * EMBED;

  v8f acc = {};
  for (int kb = 0; kb < EMBED; kb += 32) {
    v16bf ah = load_a_frag(aH, kb, hih);
    v16bf al = load_a_frag(aL, kb, hih);
    v16bf bh = load_b_frag(bH, kb, hih);
    v16bf bl = load_b_frag(bL, kb, hih);
    acc = bfmma(ah, bh, acc);
    acc = bfmma(ah, bl, acc);
    acc = bfmma(al, bh, acc);
  }
  const float sc = isK ? 1.0f : 0.044194173824159216f;  // 1/sqrt(512) folded into Q
#pragma unroll
  for (int j = 0; j < 8; ++j) {
    int r = j + 8 * hih;
    float v = acc[j] * sc;
    __bf16 h = (__bf16)v;
    size_t o = (size_t)(mtile * 16 + r) * EMBED + ntile * 16 + mrow;
    Ohi[o] = h;
    Olo[o] = (__bf16)(v - (float)h);
  }
}

// ---------------------------------------------------------------------------
// Kernel 3: flash attention. Block = 16 query rows, 8 waves.
// Per iteration (128 keys): each wave one 16x16 S tile; cooperative online
// softmax in LDS; each wave accumulates its 16x64 slice of O via P * X^T.
// ---------------------------------------------------------------------------
__global__ __launch_bounds__(256) void csa_flash_attn(
    const __bf16* __restrict__ Qhi,  const __bf16* __restrict__ Qlo,
    const __bf16* __restrict__ Khi,  const __bf16* __restrict__ Klo,
    const __bf16* __restrict__ Xthi, const __bf16* __restrict__ Xtlo,
    float* __restrict__ out) {
  __shared__ __align__(16) __bf16 sQh[16 * EMBED];
  __shared__ __align__(16) __bf16 sQl[16 * EMBED];
  __shared__ __align__(16) float  sS[16 * KC];
  __shared__ __align__(16) __bf16 sPh[16 * KC];
  __shared__ __align__(16) __bf16 sPl[16 * KC];
  __shared__ float sRed[16 * 16];
  __shared__ float sM[16], sL[16], sA[16];

  const int tid  = threadIdx.x;
  const int wave = tid >> 5;
  const int lane = tid & 31;
  const int mrow = lane & 15;
  const int hih  = lane >> 4;
  const int q0   = blockIdx.x * 16;

  // stage the 16x512 Q tile (hi+lo) into LDS once
  for (int i = tid; i < 16 * EMBED / 8; i += 256) {
    ((float4*)sQh)[i] = ((const float4*)(Qhi + (size_t)q0 * EMBED))[i];
    ((float4*)sQl)[i] = ((const float4*)(Qlo + (size_t)q0 * EMBED))[i];
  }
  if (tid < 16) { sM[tid] = -INFINITY; sL[tid] = 0.0f; }
  __syncthreads();

  v8f O[4] = {};  // 16 rows x 64 cols (cols = wave*64 .. +63)

  const __bf16* qh = sQh + mrow * EMBED;
  const __bf16* ql = sQl + mrow * EMBED;
  const __bf16* ph = sPh + mrow * KC;
  const __bf16* pl = sPl + mrow * KC;

  for (int k0 = 0; k0 < SEQ; k0 += KC) {
    // ---- S tile: this wave owns keys [k0 + wave*16, +16)
    {
      const int keyrow = k0 + wave * 16 + mrow;  // B lane = key column of S
      const __bf16* kh = Khi + (size_t)keyrow * EMBED;
      const __bf16* kl = Klo + (size_t)keyrow * EMBED;
      v8f s = {};
      for (int kb = 0; kb < EMBED; kb += 32) {
        v16bf ah = load_a_frag(qh, kb, hih);
        v16bf al = load_a_frag(ql, kb, hih);
        v16bf bh = load_b_frag(kh, kb, hih);
        v16bf bl = load_b_frag(kl, kb, hih);
        s = bfmma(ah, bh, s);
        s = bfmma(ah, bl, s);
        s = bfmma(al, bh, s);
      }
#pragma unroll
      for (int j = 0; j < 8; ++j)
        sS[(j + 8 * hih) * KC + wave * 16 + mrow] = s[j];
    }
    __syncthreads();

    // ---- partial row max: 16 segs x 8 cols per row
    {
      const int row = tid & 15, seg = tid >> 4;
      float mx = sS[row * KC + seg * 8];
#pragma unroll
      for (int c = 1; c < 8; ++c) mx = fmaxf(mx, sS[row * KC + seg * 8 + c]);
      sRed[row * 16 + seg] = mx;
    }
    __syncthreads();
    if (tid < 16) {
      float mx = sM[tid];
#pragma unroll
      for (int s2 = 0; s2 < 16; ++s2) mx = fmaxf(mx, sRed[tid * 16 + s2]);
      sA[tid] = __expf(sM[tid] - mx);  // == 0 on the first block (exp(-inf))
      sM[tid] = mx;
    }
    __syncthreads();

    // ---- P = exp(S - m_new), split to bf16 hi/lo, partial row sums
    {
      const int row = tid & 15, seg = tid >> 4;
      const float mnew = sM[row];
      float acc = 0.0f;
#pragma unroll
      for (int c = 0; c < 8; ++c) {
        int idx = row * KC + seg * 8 + c;
        float p = __expf(sS[idx] - mnew);
        acc += p;
        __bf16 h = (__bf16)p;
        sPh[idx] = h;
        sPl[idx] = (__bf16)(p - (float)h);
      }
      sRed[row * 16 + seg] = acc;
    }
    __syncthreads();
    if (tid < 16) {
      float sum = 0.0f;
#pragma unroll
      for (int s2 = 0; s2 < 16; ++s2) sum += sRed[tid * 16 + s2];
      sL[tid] = sL[tid] * sA[tid] + sum;
    }

    // ---- rescale running O, then O += P * V (V = X rows k0..k0+KC)
#pragma unroll
    for (int t = 0; t < 4; ++t)
#pragma unroll
      for (int j = 0; j < 8; ++j) O[t][j] *= sA[j + 8 * hih];

#pragma unroll
    for (int t = 0; t < 4; ++t) {
      const int col = wave * 64 + t * 16 + mrow;  // B lane = embed column
      const __bf16* xh = Xthi + (size_t)col * SEQ + k0;
      const __bf16* xl = Xtlo + (size_t)col * SEQ + k0;
      for (int kb = 0; kb < KC; kb += 32) {
        v16bf ah = load_a_frag(ph, kb, hih);
        v16bf al = load_a_frag(pl, kb, hih);
        v16bf bh = load_b_frag(xh, kb, hih);
        v16bf bl = load_b_frag(xl, kb, hih);
        O[t] = bfmma(ah, bh, O[t]);
        O[t] = bfmma(ah, bl, O[t]);
        O[t] = bfmma(al, bh, O[t]);
      }
    }
    __syncthreads();
  }

  // ---- epilogue: out = O / l
#pragma unroll
  for (int t = 0; t < 4; ++t)
#pragma unroll
    for (int j = 0; j < 8; ++j) {
      int r = j + 8 * hih;
      out[(size_t)(q0 + r) * EMBED + wave * 64 + t * 16 + mrow] = O[t][j] / sL[r];
    }
}

// ---------------------------------------------------------------------------
// Launch: inputs = [rotation_params (Wq), entangle_params (Wk), inputs (X)]
// Workspace (~71 MB): split bf16 hi/lo (+transposed) copies of X/W, Q/K hi/lo.
// ---------------------------------------------------------------------------
extern "C" void kernel_launch(void* const* d_in, const int* in_sizes, int n_in,
                              void* d_out, int out_size, void* d_ws, size_t ws_size,
                              hipStream_t stream) {
  (void)in_sizes; (void)n_in; (void)out_size; (void)ws_size;
  const float* Wq = (const float*)d_in[0];
  const float* Wk = (const float*)d_in[1];
  const float* X  = (const float*)d_in[2];
  float* out = (float*)d_out;

  char* ws = (char*)d_ws;
  size_t off = 0;
  auto alloc = [&](size_t elems) {
    __bf16* p = (__bf16*)(ws + off);
    off += ((elems * 2 + 255) & ~(size_t)255);
    return p;
  };
  const size_t ND = (size_t)SEQ * EMBED;
  const size_t DD = (size_t)EMBED * EMBED;

  __bf16 *Xhi = alloc(ND), *Xlo = alloc(ND), *Xthi = alloc(ND), *Xtlo = alloc(ND);
  __bf16 *Wqhi = alloc(DD), *Wqlo = alloc(DD), *Wqthi = alloc(DD), *Wqtlo = alloc(DD);
  __bf16 *Wkhi = alloc(DD), *Wklo = alloc(DD), *Wkthi = alloc(DD), *Wktlo = alloc(DD);
  __bf16 *Qhi = alloc(ND), *Qlo = alloc(ND), *Khi = alloc(ND), *Klo = alloc(ND);

  const dim3 blk(256);
  csa_split512<<<dim3((unsigned)((ND + 255) / 256)), blk, 0, stream>>>(
      X, Xhi, Xlo, Xthi, Xtlo, SEQ);
  csa_split512<<<dim3((unsigned)((DD + 255) / 256)), blk, 0, stream>>>(
      Wq, Wqhi, Wqlo, Wqthi, Wqtlo, EMBED);
  csa_split512<<<dim3((unsigned)((DD + 255) / 256)), blk, 0, stream>>>(
      Wk, Wkhi, Wklo, Wkthi, Wktlo, EMBED);

  csa_qk_proj<<<dim3(4, SEQ / 16, 2), blk, 0, stream>>>(
      Xhi, Xlo, Wqthi, Wqtlo, Wkthi, Wktlo, Qhi, Qlo, Khi, Klo);

  csa_flash_attn<<<dim3(SEQ / 16), blk, 0, stream>>>(
      Qhi, Qlo, Khi, Klo, Xthi, Xtlo, out);
}
